// FourLayers_58325655880087
// MI455X (gfx1250) — matrix operand order repaired and checked by reading, not compiled, not run on hardware
//
#include <hip/hip_runtime.h>
#include <hip/hip_bf16.h>

// ---------------------------------------------------------------------------
// 4-layer LSTM (B=64, T=512, IN=256, H=512) + FC(512->256), fp32 reference.
//
// MI455X strategy (gfx1250, wave32, WMMA, 320KB LDS/WGP, 23.3 TB/s HBM):
//  Phase 1/3: parallel bf16 WMMA GEMMs (xg precompute, FC), f32 accumulate,
//             async global->LDS DMA staging with LDS double buffering.
//  Phase 2:   persistent recurrent scan, 32 WGs x 128 thr; each WG keeps its
//             W_hh gate slice (64KB bf16) resident in LDS for all 512 steps.
//             Per step: each wave async-DMAs only its own 16 batch rows of h
//             (16KB) into LDS (no block barrier needed), runs a software-
//             pipelined 16-iteration k-loop of v_wmma_f32_16x16x32_bf16,
//             applies the LSTM cell with c in f32 registers, writes h (bf16)
//             to a ping-pong global buffer (fixes cross-WG WAR hazard), and
//             synchronizes timesteps with a device-scope release/acquire
//             spin barrier. xg streams exactly once (1GB ~45us @ HBM peak);
//             the scan is WMMA-latency bound on the dependency chain.
// ---------------------------------------------------------------------------

typedef __attribute__((ext_vector_type(16))) __bf16 bf16x16;
typedef __attribute__((ext_vector_type(8)))  __bf16 bf16x8;
typedef __attribute__((ext_vector_type(8)))  float  v8f;
typedef int v4i __attribute__((vector_size(16)));

#define LSTM_B   64
#define LSTM_T   512
#define LSTM_H   512
#define LSTM_4H  2048
#define SCAN_WGS 32

#if __has_builtin(__builtin_amdgcn_global_load_async_to_lds_b128) && \
    __has_builtin(__builtin_amdgcn_s_wait_asynccnt)
#define HAVE_ASYNC_LDS 1
#else
#define HAVE_ASYNC_LDS 0
#endif

#define AS_GLOBAL __attribute__((address_space(1)))
#define AS_LDS    __attribute__((address_space(3)))

__device__ __forceinline__ void copy16_g2l(const void* g, void* l) {
#if HAVE_ASYNC_LDS
    __builtin_amdgcn_global_load_async_to_lds_b128(
        (AS_GLOBAL v4i*)(g), (AS_LDS v4i*)(l), 0, 0);
#else
    *(uint4*)l = *(const uint4*)g;
#endif
}
__device__ __forceinline__ void wait_async() {
#if HAVE_ASYNC_LDS
    __builtin_amdgcn_s_wait_asynccnt(0);
#endif
}

__device__ __forceinline__ float fast_sigmoid(float x) {
    return 1.0f / (1.0f + __expf(-x));
}
__device__ __forceinline__ float fast_tanh(float x) {
    return 2.0f / (1.0f + __expf(-2.0f * x)) - 1.0f;
}

// ------------------------- f32 -> bf16 conversion --------------------------
__global__ void cvt_f32_bf16(const float* __restrict__ src,
                             __bf16* __restrict__ dst, int n) {
    int i = blockIdx.x * blockDim.x + threadIdx.x;
    int stride = gridDim.x * blockDim.x;
    for (; i < n; i += stride) dst[i] = (__bf16)src[i];
}

// --------------- zero ping-pong h-state + barrier counter ------------------
__global__ void init_state(__bf16* __restrict__ hstate, int n,
                           int* __restrict__ bar) {
    int i = blockIdx.x * blockDim.x + threadIdx.x;
    if (i == 0) *bar = 0;
    int stride = gridDim.x * blockDim.x;
    for (; i < n; i += stride) hstate[i] = (__bf16)0.0f;
}

// ---------------------------------------------------------------------------
// Tiled bf16 WMMA GEMM:  C[M,N](f32) = A[M,K](bf16) @ Bw[N,K](bf16)^T + bias
// grid = (M/64, N/64), block = 128 threads (4 waves).
// Async-DMA double-buffered LDS staging: fetch of tile k+32 overlaps the
// four WMMAs of tile k.
// ---------------------------------------------------------------------------
__global__ void gemm_bf16_bias(const __bf16* __restrict__ A,
                               const __bf16* __restrict__ Bw,
                               const float* __restrict__ bias1,
                               const float* __restrict__ bias2,
                               float* __restrict__ C, int K, int N) {
    __shared__ __bf16 At[2][64 * 32];
    __shared__ __bf16 Bt[2][64 * 32];
    const int tid  = threadIdx.x;
    const int lane = tid & 31;
    const int wave = tid >> 5;
    const int m0 = blockIdx.x * 64;
    const int n0 = blockIdx.y * 64;

    const v8f accz = {0.f,0.f,0.f,0.f,0.f,0.f,0.f,0.f};
    v8f acc0 = accz, acc1 = accz, acc2 = accz, acc3 = accz;

    const int half8 = (lane >= 16) ? 8 : 0;
    const int koffB = (lane >= 16) ? 16 : 0;
    const int nloc  = lane & 15;
    const int arow  = wave * 16 + nloc;

    // Staging: thread covers segments tid and tid+128 of the 256 16B chunks.
    const int r0 = tid >> 2, r1 = (tid + 128) >> 2;
    const int cs = (tid & 3) * 8;

    auto stage = [&](int buf, int k0) {
        copy16_g2l(&A [(size_t)(m0 + r0) * K + k0 + cs], &At[buf][r0 * 32 + cs]);
        copy16_g2l(&A [(size_t)(m0 + r1) * K + k0 + cs], &At[buf][r1 * 32 + cs]);
        copy16_g2l(&Bw[(size_t)(n0 + r0) * K + k0 + cs], &Bt[buf][r0 * 32 + cs]);
        copy16_g2l(&Bw[(size_t)(n0 + r1) * K + k0 + cs], &Bt[buf][r1 * 32 + cs]);
    };

    stage(0, 0);
    wait_async();
    __syncthreads();

    int buf = 0;
    for (int k0 = 0; k0 < K; k0 += 32) {
        if (k0 + 32 < K) stage(buf ^ 1, k0 + 32);   // overlaps WMMAs below

        bf16x8 alo = *(const bf16x8*)&At[buf][arow * 32 + koffB];
        bf16x8 ahi = *(const bf16x8*)&At[buf][arow * 32 + koffB + 8];
        bf16x16 af;
        #pragma unroll
        for (int i = 0; i < 8; ++i) { af[i] = alo[i]; af[i + 8] = ahi[i]; }

        bf16x16 b0 = *(const bf16x16*)&Bt[buf][( 0 + nloc) * 32 + koffB];
        bf16x16 b1 = *(const bf16x16*)&Bt[buf][(16 + nloc) * 32 + koffB];
        bf16x16 b2 = *(const bf16x16*)&Bt[buf][(32 + nloc) * 32 + koffB];
        bf16x16 b3 = *(const bf16x16*)&Bt[buf][(48 + nloc) * 32 + koffB];

        acc0 = __builtin_amdgcn_wmma_f32_16x16x32_bf16(false, af, false, b0,
                                                       (short)0, acc0, false, false);
        acc1 = __builtin_amdgcn_wmma_f32_16x16x32_bf16(false, af, false, b1,
                                                       (short)0, acc1, false, false);
        acc2 = __builtin_amdgcn_wmma_f32_16x16x32_bf16(false, af, false, b2,
                                                       (short)0, acc2, false, false);
        acc3 = __builtin_amdgcn_wmma_f32_16x16x32_bf16(false, af, false, b3,
                                                       (short)0, acc3, false, false);

        wait_async();
        __syncthreads();
        buf ^= 1;
    }

    const int mofs = (lane >= 16) ? 8 : 0;
    v8f accs[4] = {acc0, acc1, acc2, acc3};
    #pragma unroll
    for (int nt = 0; nt < 4; ++nt) {
        int col = n0 + nt * 16 + nloc;
        float bs = bias1 ? bias1[col] : 0.0f;
        if (bias2) bs += bias2[col];
        #pragma unroll
        for (int j = 0; j < 8; ++j) {
            int row = m0 + wave * 16 + mofs + j;
            C[(size_t)row * N + col] = accs[nt][j] + bs;
        }
    }
}

// ---------------------------------------------------------------------------
// Persistent LSTM scan. 32 WGs x 128 threads. WG bx owns h-cols
// [bx*16, bx*16+16). Dynamic LDS 128KB: [0,64K) W_hh slice, [64K,128K) h.
// hstate is ping-pong: read buf[t&1], write buf[(t+1)&1] (32768 elems each).
// ---------------------------------------------------------------------------
__global__ void lstm_scan(const float* __restrict__ xg,       // [B][T][4H] f32
                          const __bf16* __restrict__ whh,     // [4H][H] bf16
                          __bf16* __restrict__ hstate,        // 2x[B][H] bf16
                          __bf16* __restrict__ hs_out,        // [B][T][H] bf16
                          int* __restrict__ bar) {
    extern __shared__ char smem[];
    __bf16* wslice = (__bf16*)smem;                // [64][512]
    __bf16* hbuf   = (__bf16*)(smem + 65536);      // [64][512]

    const int tid  = threadIdx.x;
    const int lane = tid & 31;
    const int wave = tid >> 5;
    const int c0   = blockIdx.x * 16;              // h-column base

    const int mbase = wave * 16;
    const int nloc  = lane & 15;
    const int mofs  = (lane >= 16) ? 8 : 0;
    const int half8 = (lane >= 16) ? 8 : 0;
    const int koffB = (lane >= 16) ? 16 : 0;
    const int arow  = mbase + nloc;
    const int col   = c0 + nloc;

    // Preload W_hh slice once: local row g*16+r <- global row g*512+c0+r.
    #pragma unroll 4
    for (int j = 0; j < 32; ++j) {
        int u = tid + j * 128;
        int lr = u >> 5, ks = u & 31;
        int gate = lr >> 4, r = lr & 15;
        copy16_g2l(&whh[(size_t)(gate * 512 + c0 + r) * 512 + ks * 8],
                   &wslice[lr * 512 + ks * 8]);
    }
    wait_async();
    __syncthreads();

    const v8f vz = {0.f,0.f,0.f,0.f,0.f,0.f,0.f,0.f};
    v8f cst = vz;                                  // cell state, f32 regs

    // Fragment loaders (16x16x32 bf16, contiguous k order: k = 16*(lane/16) + s).
    auto loadA = [&](int k0) {
        bf16x8 lo = *(const bf16x8*)&hbuf[arow * 512 + k0 + koffB];
        bf16x8 hi = *(const bf16x8*)&hbuf[arow * 512 + k0 + koffB + 8];
        bf16x16 af;
        #pragma unroll
        for (int i = 0; i < 8; ++i) { af[i] = lo[i]; af[i + 8] = hi[i]; }
        return af;
    };
    auto loadB = [&](int g, int k0) {
        return *(const bf16x16*)&wslice[(g * 16 + nloc) * 512 + k0 + koffB];
    };

    for (int t = 0; t < LSTM_T; ++t) {
        const __bf16* hsrc = hstate + (size_t)(t & 1) * (LSTM_B * LSTM_H);
        __bf16*       hdst = hstate + (size_t)((t + 1) & 1) * (LSTM_B * LSTM_H);

        // Issue xg gate loads for this step early (overlap with h-staging
        // and the WMMA k-loop; compiler inserts loadcnt waits at first use).
        v8f xI, xF, xG, xO;
        #pragma unroll
        for (int j = 0; j < 8; ++j) {
            int b = mbase + mofs + j;
            size_t base = ((size_t)b * LSTM_T + t) * LSTM_4H;
            xI[j] = xg[base +        col];
            xF[j] = xg[base +  512 + col];
            xG[j] = xg[base + 1024 + col];
            xO[j] = xg[base + 1536 + col];
        }

        // Per-wave h staging: this wave's A-fragments only read its own 16
        // batch rows, so each wave DMAs exactly those rows (16KB) and waits
        // only on its own ASYNCcnt -- no block barrier needed.
        #pragma unroll 8
        for (int q = 0; q < 32; ++q) {
            int u = lane + q * 32;          // 0..1023 16B units
            int r = u >> 6;                 // row within wave tile (0..15)
            int off = (u & 63) * 8;         // bf16 element offset in row
            copy16_g2l(&hsrc[(size_t)(mbase + r) * 512 + off],
                       &hbuf[(mbase + r) * 512 + off]);
        }
        wait_async();

        // Software-pipelined k-loop: load k+32 fragments while issuing the
        // four WMMAs for k. Fully unrolled (16 iterations).
        v8f aI = vz, aF = vz, aG = vz, aO = vz;
        bf16x16 ca = loadA(0);
        bf16x16 cI = loadB(0, 0), cF = loadB(1, 0),
                cG = loadB(2, 0), cO = loadB(3, 0);
        #pragma unroll
        for (int k0 = 0; k0 < LSTM_H; k0 += 32) {
            bf16x16 na, nI, nF, nG, nO;
            const bool more = (k0 + 32 < LSTM_H);
            if (more) {
                na = loadA(k0 + 32);
                nI = loadB(0, k0 + 32); nF = loadB(1, k0 + 32);
                nG = loadB(2, k0 + 32); nO = loadB(3, k0 + 32);
            }
            aI = __builtin_amdgcn_wmma_f32_16x16x32_bf16(false, ca, false, cI,
                                                         (short)0, aI, false, false);
            aF = __builtin_amdgcn_wmma_f32_16x16x32_bf16(false, ca, false, cF,
                                                         (short)0, aF, false, false);
            aG = __builtin_amdgcn_wmma_f32_16x16x32_bf16(false, ca, false, cG,
                                                         (short)0, aG, false, false);
            aO = __builtin_amdgcn_wmma_f32_16x16x32_bf16(false, ca, false, cO,
                                                         (short)0, aO, false, false);
            if (more) { ca = na; cI = nI; cF = nF; cG = nG; cO = nO; }
        }

        // Elementwise LSTM cell for this wave's 16x16 (batch x h-col) block.
        #pragma unroll
        for (int j = 0; j < 8; ++j) {
            int b = mbase + mofs + j;
            float gi = fast_sigmoid(xI[j] + aI[j]);
            float gf = fast_sigmoid(xF[j] + aF[j]);
            float gg = fast_tanh   (xG[j] + aG[j]);
            float go = fast_sigmoid(xO[j] + aO[j]);
            float cv = gf * cst[j] + gi * gg;
            cst[j] = cv;
            float h = go * fast_tanh(cv);
            __bf16 hb = (__bf16)h;
            hdst[(size_t)b * LSTM_H + col] = hb;
            hs_out[((size_t)b * LSTM_T + t) * LSTM_H + col] = hb;
        }

        // Prefetch next timestep's xg slice (global_prefetch_b8).
        if (t + 1 < LSTM_T)
            __builtin_prefetch(&xg[((size_t)(mbase + mofs) * LSTM_T + t + 1) * LSTM_4H + col], 0, 1);

        // Device-scope release/acquire grid barrier between timesteps.
        __threadfence();
        __syncthreads();
        if (tid == 0) {
            __hip_atomic_fetch_add(bar, 1, __ATOMIC_RELEASE,
                                   __HIP_MEMORY_SCOPE_AGENT);
            const int target = SCAN_WGS * (t + 1);
            while (__hip_atomic_load(bar, __ATOMIC_ACQUIRE,
                                     __HIP_MEMORY_SCOPE_AGENT) < target) {
                __builtin_amdgcn_s_sleep(2);
            }
        }
        __syncthreads();
    }
}

// ---------------------------------------------------------------------------
extern "C" void kernel_launch(void* const* d_in, const int* in_sizes, int n_in,
                              void* d_out, int out_size, void* d_ws, size_t ws_size,
                              hipStream_t stream) {
    (void)in_sizes; (void)n_in; (void)out_size; (void)ws_size;
    const float* x    = (const float*)d_in[0];
    const float* w_ih[4] = {(const float*)d_in[1],  (const float*)d_in[5],
                            (const float*)d_in[9],  (const float*)d_in[13]};
    const float* w_hh[4] = {(const float*)d_in[2],  (const float*)d_in[6],
                            (const float*)d_in[10], (const float*)d_in[14]};
    const float* b_ih[4] = {(const float*)d_in[3],  (const float*)d_in[7],
                            (const float*)d_in[11], (const float*)d_in[15]};
    const float* b_hh[4] = {(const float*)d_in[4],  (const float*)d_in[8],
                            (const float*)d_in[12], (const float*)d_in[16]};
    const float* fc_w = (const float*)d_in[17];
    const float* fc_b = (const float*)d_in[18];
    float* out = (float*)d_out;

    // Workspace layout (all sizes multiples of 256B).
    char* ws = (char*)d_ws;
    const size_t SZ_XG   = (size_t)LSTM_B * LSTM_T * LSTM_4H * 4;  // 256MB f32
    const size_t SZ_ACT  = (size_t)LSTM_B * LSTM_T * LSTM_H  * 2;  // 32MB bf16
    const size_t SZ_W    = (size_t)LSTM_4H * LSTM_H * 2;           // 2MB bf16
    const size_t SZ_FCW  = (size_t)256 * LSTM_H * 2;
    const size_t SZ_HST  = (size_t)2 * LSTM_B * LSTM_H * 2;        // ping-pong

    float*  xg     = (float*)ws;                          ws += SZ_XG;
    __bf16* actA   = (__bf16*)ws;                         ws += SZ_ACT;
    __bf16* actB   = (__bf16*)ws;                         ws += SZ_ACT;
    __bf16* wih_bf[4], *whh_bf[4];
    for (int l = 0; l < 4; ++l) { wih_bf[l] = (__bf16*)ws; ws += SZ_W; }
    for (int l = 0; l < 4; ++l) { whh_bf[l] = (__bf16*)ws; ws += SZ_W; }
    __bf16* fcw_bf = (__bf16*)ws;                         ws += SZ_FCW;
    __bf16* hstate = (__bf16*)ws;                         ws += SZ_HST;
    int*    bar    = (int*)ws;                            ws += 256;

    // ---- Convert inputs/weights to bf16 once per call ----
    cvt_f32_bf16<<<2048, 256, 0, stream>>>(x, actA, LSTM_B * LSTM_T * 256);
    for (int l = 0; l < 4; ++l) {
        int din = (l == 0) ? 256 : LSTM_H;
        cvt_f32_bf16<<<1024, 256, 0, stream>>>(w_ih[l], wih_bf[l], LSTM_4H * din);
        cvt_f32_bf16<<<1024, 256, 0, stream>>>(w_hh[l], whh_bf[l], LSTM_4H * LSTM_H);
    }
    cvt_f32_bf16<<<256, 256, 0, stream>>>(fc_w, fcw_bf, 256 * LSTM_H);

    // ---- 4 LSTM layers ----
    const int M = LSTM_B * LSTM_T;  // 32768
    for (int l = 0; l < 4; ++l) {
        int din = (l == 0) ? 256 : LSTM_H;
        const __bf16* in_act  = (l & 1) ? actB : actA;
        __bf16*       out_act = (l & 1) ? actA : actB;

        // Phase 1: xg = in_act @ w_ih^T + (b_ih + b_hh)
        dim3 g1(M / 64, LSTM_4H / 64);
        gemm_bf16_bias<<<g1, 128, 0, stream>>>(in_act, wih_bf[l],
                                               b_ih[l], b_hh[l],
                                               xg, din, LSTM_4H);

        // Phase 2: persistent recurrent scan (zero both h ping-pong bufs)
        init_state<<<64, 256, 0, stream>>>(hstate, 2 * LSTM_B * LSTM_H, bar);
        lstm_scan<<<SCAN_WGS, 128, 131072, stream>>>(xg, whh_bf[l],
                                                     hstate, out_act, bar);
    }

    // ---- FC: out[b,t,:] = h4 @ fc_w^T + fc_b ----
    const __bf16* h4 = actA;  // layer 3 wrote actA
    dim3 gf(M / 64, 256 / 64);
    gemm_bf16_bias<<<gf, 128, 0, stream>>>(h4, fcw_bf, fc_b, nullptr,
                                           out, LSTM_H, 256);
}
